// GroupedQueryAttention_66254165508267
// MI455X (gfx1250) — compile-verified, matrix operand order checked by reading, not run on hardware
//
#include <hip/hip_runtime.h>
#include <hip/hip_bf16.h>
#include <stdint.h>

// ---------------------------------------------------------------------------
// GQA forward for MI455X (gfx1250, wave32, WMMA, async global->LDS DMA).
// Pipeline: fp32->bf16 convert (weights transposed to K-major),
// Q/K/V projection GEMMs (bf16 WMMA, f32 acc, double-buffered async staging;
// V emitted dim-major), flash-attention GQA core (double-buffered async K/V
// tiles, online softmax), output projection GEMM (f32 out + bias).
// All LDS tiles are K-major for their consumer so every WMMA operand fragment
// is exactly two ds_load_b128.
// ---------------------------------------------------------------------------

typedef __attribute__((ext_vector_type(16))) __bf16        v16bf;
typedef __attribute__((ext_vector_type(8)))  float         v8f;
typedef __attribute__((ext_vector_type(4)))  unsigned int  v4u;

__device__ __forceinline__ unsigned short f2bf_bits(float f) {
    unsigned int u = __float_as_uint(f);
    unsigned int r = u + 0x7FFFu + ((u >> 16) & 1u);   // round-to-nearest-even
    return (unsigned short)(r >> 16);
}

union Frag16 { v16bf v; unsigned int u[8]; };

// Load a 16x32 bf16 WMMA operand fragment (A/B layout, 7.12.2) from LDS.
__device__ __forceinline__ void load_frag_lds(Frag16& f,
                                              const unsigned int* lds,
                                              int base) {
    *(v4u*)&f.u[0] = *(const v4u*)(lds + base);
    *(v4u*)&f.u[4] = *(const v4u*)(lds + base + 8);
}

// Async global->LDS 16-byte copy (per-lane), tracked by ASYNCcnt.
__device__ __forceinline__ void async_b128(const unsigned int* gsrc,
                                           const unsigned int* ldst) {
    unsigned lds_off = (unsigned)(unsigned long long)(const void*)ldst;
    unsigned long long gaddr = (unsigned long long)(const void*)gsrc;
    asm volatile("global_load_async_to_lds_b128 %0, %1, off"
                 :: "v"(lds_off), "v"(gaddr) : "memory");
}
__device__ __forceinline__ void wait_async0() {
    asm volatile("s_wait_asynccnt 0x0" ::: "memory");
}
__device__ __forceinline__ void wait_async3() {
    asm volatile("s_wait_asynccnt 0x3" ::: "memory");
}
__device__ __forceinline__ void wait_async4() {
    asm volatile("s_wait_asynccnt 0x4" ::: "memory");
}

// ---------------------------------------------------------------------------
// fp32 -> bf16 converts (plain and transposing: out[c*R + r] = in[r*C + c])
// ---------------------------------------------------------------------------
__global__ void f32_to_bf16_kernel(const float* __restrict__ in,
                                   unsigned short* __restrict__ out, int n) {
    for (int i = blockIdx.x * blockDim.x + threadIdx.x; i < n;
         i += gridDim.x * blockDim.x)
        out[i] = f2bf_bits(in[i]);
}

__global__ void f32_to_bf16_t_kernel(const float* __restrict__ in,
                                     unsigned short* __restrict__ out, int R,
                                     int C) {
    for (int i = blockIdx.x * blockDim.x + threadIdx.x; i < R * C;
         i += gridDim.x * blockDim.x) {
        int r = i / C, c = i - r * C;
        out[(size_t)c * R + r] = f2bf_bits(in[i]);
    }
}

// ---------------------------------------------------------------------------
// Tiled bf16 GEMM:  O[M,N] = A[M,K] (bf16) @ Wt[N,K]^T (bf16) + bias
// Wt is K-major (pre-transposed weights). Block = 256 threads (8 waves),
// tile 128x64, BK=32, double-buffered async global->LDS staging.
// storeT=1 writes the bf16 result transposed: O[N,M] (dim-major V).
// ---------------------------------------------------------------------------
__global__ __launch_bounds__(256) void gemm_bf16_wmma(
    const unsigned short* __restrict__ A, const unsigned short* __restrict__ Wt,
    const float* __restrict__ bias, unsigned short* __restrict__ obf,
    float* __restrict__ of32, int M, int N, int K, int storeF32, int storeT) {
    __shared__ unsigned int sAd[2][128 * 20];  // 128 rows x 32 bf16 (80B rows)
    __shared__ unsigned int sBd[2][64 * 20];   // 64 n-rows x 32 bf16 (K-major)

    const int tid = threadIdx.x, lane = tid & 31, wv = tid >> 5;
    const int half = lane >> 4, l16 = lane & 15;
    const int mw = wv & 3, nw = wv >> 2;
    const int rowBase = blockIdx.y * 128, nBase = blockIdx.x * 64;
    const unsigned int* Ad = reinterpret_cast<const unsigned int*>(A);
    const unsigned int* Wd = reinterpret_cast<const unsigned int*>(Wt);
    const int Kd2 = K >> 1;

    const v8f vzero = {0.f, 0.f, 0.f, 0.f, 0.f, 0.f, 0.f, 0.f};
    v8f acc[2][2];
#pragma unroll
    for (int mi = 0; mi < 2; ++mi)
#pragma unroll
        for (int ni = 0; ni < 2; ++ni) acc[mi][ni] = vzero;

    const int nk = K >> 5;
    auto issue_tiles = [&](int kt, int buf) {
        const int kB2 = kt << 4;  // kBase/2 in dwords
        // A tile: 512 b128 chunks -> 2 per thread
#pragma unroll
        for (int i = 0; i < 2; ++i) {
            int d = tid + (i << 8), r = d >> 2, c4 = d & 3;
            async_b128(Ad + (size_t)(rowBase + r) * Kd2 + kB2 + c4 * 4,
                       &sAd[buf][r * 20 + c4 * 4]);
        }
        // B tile (K-major): 256 b128 chunks -> 1 per thread
        {
            int n = tid >> 2, c4 = tid & 3;
            async_b128(Wd + (size_t)(nBase + n) * Kd2 + kB2 + c4 * 4,
                       &sBd[buf][n * 20 + c4 * 4]);
        }
    };

    issue_tiles(0, 0);
    for (int kt = 0; kt < nk; ++kt) {
        if (kt + 1 < nk) {
            issue_tiles(kt + 1, (kt + 1) & 1);
            wait_async3();  // previous tile's 3 async ops complete (in-order)
        } else {
            wait_async0();
        }
        __syncthreads();

        const unsigned int* sA = sAd[kt & 1];
        const unsigned int* sB = sBd[kt & 1];
        Frag16 af[2], bfr[2];
#pragma unroll
        for (int mi = 0; mi < 2; ++mi)
            load_frag_lds(af[mi], sA, (mw * 32 + mi * 16 + l16) * 20 + half * 4);
#pragma unroll
        for (int ni = 0; ni < 2; ++ni)
            load_frag_lds(bfr[ni], sB, (nw * 32 + ni * 16 + l16) * 20 + half * 4);
#pragma unroll
        for (int mi = 0; mi < 2; ++mi)
#pragma unroll
            for (int ni = 0; ni < 2; ++ni)
                acc[mi][ni] = __builtin_amdgcn_wmma_f32_16x16x32_bf16(
                    false, af[mi].v, false, bfr[ni].v, (short)0, acc[mi][ni],
                    false, false);
        __syncthreads();
    }

    // Epilogue: C/D layout -> global
#pragma unroll
    for (int mi = 0; mi < 2; ++mi)
#pragma unroll
        for (int ni = 0; ni < 2; ++ni)
#pragma unroll
            for (int r = 0; r < 8; ++r) {
                int grow = rowBase + mw * 32 + mi * 16 + half * 8 + r;
                int gcol = nBase + nw * 32 + ni * 16 + l16;
                float v = acc[mi][ni][r] + bias[gcol];
                if (storeF32)
                    of32[(size_t)grow * N + gcol] = v;
                else if (storeT)
                    obf[(size_t)gcol * M + grow] = f2bf_bits(v);
                else
                    obf[(size_t)grow * N + gcol] = f2bf_bits(v);
            }
}

// ---------------------------------------------------------------------------
// Flash-attention GQA core.
// Grid: (S/128, H, B). Block = 256 threads = 8 waves; wave w owns query rows
// [w*16, w*16+16). 64-key K (key-major) and Vt (dim-major) tiles stream via
// double-buffered async global->LDS DMA. Online softmax per query row.
// ---------------------------------------------------------------------------
__global__ __launch_bounds__(256) void gqa_attn_wmma(
    const unsigned short* __restrict__ Q, const unsigned short* __restrict__ Kc,
    const unsigned short* __restrict__ Vt, unsigned short* __restrict__ C,
    int S, int M) {
    __shared__ unsigned int sKd[2][64 * 40];   // key-major: 64 keys x 64 dim
    __shared__ unsigned int sVtd[2][64 * 40];  // dim-major: 64 dim x 64 keys
    __shared__ unsigned int sPd[128 * 40];     // per-wave 16x64 P (key-major)
    unsigned short* sP16 = reinterpret_cast<unsigned short*>(sPd);

    const int tid = threadIdx.x, lane = tid & 31, wv = tid >> 5;
    const int half = lane >> 4, l16 = lane & 15;
    const int h = blockIdx.y, b = blockIdx.z, g = h >> 2;  // GS = 4
    const int q0 = blockIdx.x * 128;
    const unsigned int* Qd  = reinterpret_cast<const unsigned int*>(Q);
    const unsigned int* Kd  = reinterpret_cast<const unsigned int*>(Kc);
    const unsigned int* Vtd = reinterpret_cast<const unsigned int*>(Vt);
    const int Md2 = M >> 1;

    // Q fragments for this wave's 16 rows (16x64 = two 16x32 A operands).
    Frag16 qf[2];
    const int qrow = b * S + q0 + wv * 16 + l16;
#pragma unroll
    for (int f = 0; f < 2; ++f) {
        const unsigned int* gq =
            Qd + (size_t)qrow * 1024 + (h << 5) + f * 16 + half * 4;
        *(v4u*)&qf[f].u[0] = *(const v4u*)gq;
        *(v4u*)&qf[f].u[4] = *(const v4u*)(gq + 8);
    }

    const v8f vzero = {0.f, 0.f, 0.f, 0.f, 0.f, 0.f, 0.f, 0.f};
    v8f acc[4];
#pragma unroll
    for (int dt = 0; dt < 4; ++dt) acc[dt] = vzero;
    float mrow[8], lrow[8];
#pragma unroll
    for (int r = 0; r < 8; ++r) { mrow[r] = -3.0e38f; lrow[r] = 0.f; }
    const float scale = 0.125f;  // 1/sqrt(64)

    const int nkt = S >> 6;
    auto issue_kv = [&](int kt, int buf) {
        // K tile: 64 keys x 32 dwords dim (2 b128 chunks / thread)
#pragma unroll
        for (int i = 0; i < 2; ++i) {
            int d = tid + (i << 8), r = d >> 3, c4 = d & 7;
            async_b128(Kd + (size_t)(b * S + (kt << 6) + r) * 256 + (g << 5) +
                           c4 * 4,
                       &sKd[buf][r * 40 + c4 * 4]);
        }
        // Vt tile: 64 dim rows x 32 dwords keys (2 b128 chunks / thread)
#pragma unroll
        for (int i = 0; i < 2; ++i) {
            int d = tid + (i << 8), r = d >> 3, c4 = d & 7;
            async_b128(Vtd + (size_t)((g << 6) + r) * Md2 +
                           ((b * S + (kt << 6)) >> 1) + c4 * 4,
                       &sVtd[buf][r * 40 + c4 * 4]);
        }
    };

    issue_kv(0, 0);
    for (int kt = 0; kt < nkt; ++kt) {
        if (kt + 1 < nkt) {
            issue_kv(kt + 1, (kt + 1) & 1);
            wait_async4();  // previous tile's 4 async ops complete (in-order)
        } else {
            wait_async0();
        }
        __syncthreads();
        const unsigned int* sK = sKd[kt & 1];
        const unsigned int* sV = sVtd[kt & 1];

        // scores = Q @ K^T : 4 key-tiles x 2 k-steps over head_dim
        v8f sc[4];
#pragma unroll
        for (int nt = 0; nt < 4; ++nt) sc[nt] = vzero;
#pragma unroll
        for (int nt = 0; nt < 4; ++nt) {
            int t = nt * 16 + l16;  // key index = B column
#pragma unroll
            for (int f = 0; f < 2; ++f) {
                Frag16 bk;
                load_frag_lds(bk, sK, t * 40 + f * 16 + half * 4);
                sc[nt] = __builtin_amdgcn_wmma_f32_16x16x32_bf16(
                    false, qf[f].v, false, bk.v, (short)0, sc[nt], false,
                    false);
            }
        }

        // Online softmax (per query row; 16-lane shuffle reductions)
#pragma unroll
        for (int r = 0; r < 8; ++r) {
            float sv[4];
#pragma unroll
            for (int nt = 0; nt < 4; ++nt) sv[nt] = sc[nt][r] * scale;
            float mt = fmaxf(fmaxf(sv[0], sv[1]), fmaxf(sv[2], sv[3]));
#pragma unroll
            for (int off = 1; off < 16; off <<= 1)
                mt = fmaxf(mt, __shfl_xor(mt, off, 32));
            float mnew = fmaxf(mrow[r], mt);
            float alpha = __expf(mrow[r] - mnew);
            float ls = 0.f;
            int prow = wv * 16 + half * 8 + r;
#pragma unroll
            for (int nt = 0; nt < 4; ++nt) {
                float p = __expf(sv[nt] - mnew);
                ls += p;
                sP16[prow * 80 + nt * 16 + l16] = f2bf_bits(p);
            }
#pragma unroll
            for (int off = 1; off < 16; off <<= 1)
                ls += __shfl_xor(ls, off, 32);
            lrow[r] = lrow[r] * alpha + ls;
            mrow[r] = mnew;
#pragma unroll
            for (int dt = 0; dt < 4; ++dt) acc[dt][r] = acc[dt][r] * alpha;
        }

        // P (own-wave LDS region) -> A fragments; ctx += P @ V
        Frag16 pf[2];
#pragma unroll
        for (int f = 0; f < 2; ++f)
            load_frag_lds(pf[f], sPd,
                          (wv * 16 + l16) * 40 + f * 16 + half * 4);
#pragma unroll
        for (int dt = 0; dt < 4; ++dt) {
            int n = dt * 16 + l16;  // output dim = B column
#pragma unroll
            for (int f = 0; f < 2; ++f) {
                Frag16 bv;
                load_frag_lds(bv, sV, n * 40 + f * 16 + half * 4);
                acc[dt] = __builtin_amdgcn_wmma_f32_16x16x32_bf16(
                    false, pf[f].v, false, bv.v, (short)0, acc[dt], false,
                    false);
            }
        }
        __syncthreads();
    }

    // ctx / l -> bf16 context buffer [B*S, H*64]
#pragma unroll
    for (int dt = 0; dt < 4; ++dt)
#pragma unroll
        for (int r = 0; r < 8; ++r) {
            int grow = b * S + q0 + wv * 16 + half * 8 + r;
            int gcol = (h << 6) + dt * 16 + l16;
            C[(size_t)grow * 2048 + gcol] = f2bf_bits(acc[dt][r] / lrow[r]);
        }
}

// ---------------------------------------------------------------------------
// Host launcher
// ---------------------------------------------------------------------------
extern "C" void kernel_launch(void* const* d_in, const int* in_sizes, int n_in,
                              void* d_out, int out_size, void* d_ws,
                              size_t ws_size, hipStream_t stream) {
    (void)in_sizes; (void)n_in; (void)out_size; (void)ws_size;
    const int B = 2, S = 2048, DM = 2048, H = 32, G = 8;
    const int M = B * S;                 // 4096
    const int NKV = G * 64;              // 512

    const float* x  = (const float*)d_in[0];
    const float* Wq = (const float*)d_in[1];
    const float* bq = (const float*)d_in[2];
    const float* Wk = (const float*)d_in[3];
    const float* bk = (const float*)d_in[4];
    const float* Wv = (const float*)d_in[5];
    const float* bv = (const float*)d_in[6];
    const float* Wo = (const float*)d_in[7];
    const float* bo = (const float*)d_in[8];
    float* out = (float*)d_out;

    char* ws = (char*)d_ws;
    size_t off = 0;
    auto take = [&](size_t bytes) {
        size_t o = off;
        off += (bytes + 255) & ~(size_t)255;
        return o;
    };
    unsigned short* xb  = (unsigned short*)(ws + take((size_t)M * DM * 2));
    unsigned short* wqb = (unsigned short*)(ws + take((size_t)DM * DM * 2));
    unsigned short* wkb = (unsigned short*)(ws + take((size_t)DM * NKV * 2));
    unsigned short* wvb = (unsigned short*)(ws + take((size_t)DM * NKV * 2));
    unsigned short* wob = (unsigned short*)(ws + take((size_t)DM * DM * 2));
    unsigned short* qb  = (unsigned short*)(ws + take((size_t)M * DM * 2));
    unsigned short* kb  = (unsigned short*)(ws + take((size_t)M * NKV * 2));
    unsigned short* vtb = (unsigned short*)(ws + take((size_t)M * NKV * 2));
    unsigned short* cb  = (unsigned short*)(ws + take((size_t)M * DM * 2));

    // 1) fp32 -> bf16 converts (weights transposed to K-major)
    f32_to_bf16_kernel<<<2048, 256, 0, stream>>>(x, xb, M * DM);
    f32_to_bf16_t_kernel<<<2048, 256, 0, stream>>>(Wq, wqb, DM, DM);
    f32_to_bf16_t_kernel<<<1024, 256, 0, stream>>>(Wk, wkb, DM, NKV);
    f32_to_bf16_t_kernel<<<1024, 256, 0, stream>>>(Wv, wvb, DM, NKV);
    f32_to_bf16_t_kernel<<<2048, 256, 0, stream>>>(Wo, wob, DM, DM);

    // 2) projections (bf16 out; V emitted dim-major / transposed)
    gemm_bf16_wmma<<<dim3(DM / 64, M / 128), 256, 0, stream>>>(
        xb, wqb, bq, qb, nullptr, M, DM, DM, 0, 0);
    gemm_bf16_wmma<<<dim3(NKV / 64, M / 128), 256, 0, stream>>>(
        xb, wkb, bk, kb, nullptr, M, NKV, DM, 0, 0);
    gemm_bf16_wmma<<<dim3(NKV / 64, M / 128), 256, 0, stream>>>(
        xb, wvb, bv, vtb, nullptr, M, NKV, DM, 0, 1);

    // 3) flash GQA attention (async double-buffered K/V tiles)
    gqa_attn_wmma<<<dim3(S / 128, H, B), 256, 0, stream>>>(qb, kb, vtb, cb, S,
                                                           M);

    // 4) output projection (fp32 out + bias)
    gemm_bf16_wmma<<<dim3(DM / 64, M / 128), 256, 0, stream>>>(
        cb, wob, bo, nullptr, out, M, DM, DM, 1, 0);
}
